// DeformTransformerBlock2D_82437602280008
// MI455X (gfx1250) — matrix operand order repaired and checked by reading, not compiled, not run on hardware
//
#include <hip/hip_runtime.h>
#include <math.h>

// ---------------- problem constants ----------------
#define B_   2
#define C_   256
#define H_   80
#define W_   160
#define HW_  (H_ * W_)          // 12800
#define N_   (B_ * HW_)         // 25600 tokens
#define G_   8
#define P_   8
#define CG_  (C_ / G_)          // 32
#define EC_  (2 * C_)           // 512
#define LN_EPS 1e-5f

// ---------------- WMMA vector types ----------------
typedef __attribute__((ext_vector_type(16))) _Float16 v16h;
typedef __attribute__((ext_vector_type(8)))  _Float16 v8h;
typedef __attribute__((ext_vector_type(8)))  float    v8f;

// =====================================================================
// Kernel 1: q = (feats + feats_pos) transposed NCHW -> (N, C), fp32 + f16
// =====================================================================
__global__ __launch_bounds__(256) void prep_q_kernel(
    const float* __restrict__ f, const float* __restrict__ fp,
    float* __restrict__ q32, _Float16* __restrict__ q16)
{
    __shared__ float tile[32][33];
    const int tx = threadIdx.x;          // 0..31 (spatial on load, channel on store)
    const int ty = threadIdx.y;          // 0..7
    const int s0 = blockIdx.x * 32;
    const int c0 = blockIdx.y * 32;
    const int b  = blockIdx.z;
    const size_t CHW = (size_t)C_ * HW_;

#pragma unroll
    for (int k = 0; k < 4; ++k) {
        int c = c0 + ty + 8 * k;
        size_t idx = (size_t)b * CHW + (size_t)c * HW_ + (size_t)(s0 + tx);
        tile[ty + 8 * k][tx] = f[idx] + fp[idx];     // tile[c_local][s_local]
    }
    __syncthreads();
#pragma unroll
    for (int k = 0; k < 4; ++k) {
        int s = s0 + ty + 8 * k;
        size_t n = (size_t)b * HW_ + s;
        float v = tile[tx][ty + 8 * k];              // c_local = tx
        q32[n * C_ + c0 + tx] = v;
        q16[n * C_ + c0 + tx] = (_Float16)v;
    }
}

// =====================================================================
// Kernel 2: pack weight (K x M, optional transpose) into WMMA B tiles.
// Tile (kt, nt) = 32 lanes x 16 halves contiguous, matching the
// contiguous k order the A fragment also uses: lane L -> col = L&15, K = 16*(L>>4)+i.
// =====================================================================
__global__ __launch_bounds__(256) void pack_w_kernel(
    const float* __restrict__ W, _Float16* __restrict__ out,
    int K, int M, int transp)
{
    int t = blockIdx.x * blockDim.x + threadIdx.x;
    int total = (K >> 5) * (M >> 4) * 32;
    if (t >= total) return;
    int lane = t & 31;
    int tile = t >> 5;
    int ntM  = M >> 4;
    int nt   = tile % ntM;
    int kt   = tile / ntM;
    int col  = nt * 16 + (lane & 15);
    int kbase = kt * 32 + (lane >> 4) * 16;
    _Float16* dst = out + (size_t)t * 16;
#pragma unroll
    for (int i = 0; i < 16; ++i) {
        int k = kbase + i;
        float v = transp ? W[(size_t)col * K + k] : W[(size_t)k * M + col];
        dst[i] = (_Float16)v;
    }
}

// =====================================================================
// Kernel 3: WMMA f16 GEMM:  out(N x M) = A(N x K) * Bpacked + bias
//   act == 0 : store fp32
//   act == 1 : SiLU, store f16
// Block = 256 threads = 8 waves; tile = 128 rows x 64 cols; BK = 32.
// A staged in LDS, padded stride 40 halves (80B rows -> conflict-free
// 16B ds_loads for the A fragment).
// =====================================================================
#define GEMM_BM 128
#define GEMM_BN 64
#define GEMM_LDA 40

__global__ __launch_bounds__(256) void gemm_f16_kernel(
    const _Float16* __restrict__ A, const _Float16* __restrict__ Bp,
    const float* __restrict__ bias,
    float* __restrict__ out32, _Float16* __restrict__ out16,
    int Nrows, int K, int M, int act)
{
    __shared__ _Float16 As[GEMM_BM * GEMM_LDA];

    const int tid  = threadIdx.x;
    const int wave = tid >> 5;
    const int lane = tid & 31;
    const int half = lane >> 4;
    const int l16  = lane & 15;
    const int rowBlk = blockIdx.x * GEMM_BM;
    const int colBlk = blockIdx.y * GEMM_BN;
    const int ntM = M >> 4;

    v8f acc[4] = {};
    const int nKt = K >> 5;

    for (int kt = 0; kt < nKt; ++kt) {
        // ---- cooperative A load: each thread moves 16 halves (32B) ----
        {
            int r  = tid >> 1;
            int ch = (tid & 1) * 16;
            const v8h* src = (const v8h*)(A + (size_t)(rowBlk + r) * K + kt * 32 + ch);
            v8h x0 = src[0];
            v8h x1 = src[1];
            *(v8h*)(&As[r * GEMM_LDA + ch])     = x0;
            *(v8h*)(&As[r * GEMM_LDA + ch + 8]) = x1;
            if (kt + 1 < nKt)   // lowers to global_prefetch_b8
                __builtin_prefetch(A + (size_t)(rowBlk + r) * K + (kt + 1) * 32 + ch, 0, 1);
        }
        __syncthreads();

        // ---- A fragment (16x32 f16): lane L<16 -> M=L, K 0..15 ; L>=16 -> K 16..31 ----
        const _Float16* arow = &As[(wave * 16 + l16) * GEMM_LDA + half * 16];
        v8h alo = *(const v8h*)(arow);
        v8h ahi = *(const v8h*)(arow + 8);
        v16h afrag = __builtin_shufflevector(alo, ahi,
            0, 1, 2, 3, 4, 5, 6, 7, 8, 9, 10, 11, 12, 13, 14, 15);

        // ---- B fragments from pre-swizzled tiles (32B contiguous/lane) ----
        const _Float16* bb = Bp + ((size_t)kt * ntM + (colBlk >> 4)) * 512
                                + (size_t)lane * 16;
#pragma unroll
        for (int j = 0; j < 4; ++j) {
            v16h bfrag = *(const v16h*)(bb + (size_t)j * 512);
            acc[j] = __builtin_amdgcn_wmma_f32_16x16x32_f16(
                false, afrag, false, bfrag, (short)0, acc[j], false, false);
        }
        __syncthreads();
    }

    // ---- epilogue: D layout lane<16 -> N=lane, M=r ; lane>=16 -> M=8+r ----
    const int rtop = rowBlk + wave * 16 + half * 8;
#pragma unroll
    for (int j = 0; j < 4; ++j) {
        int col = colBlk + j * 16 + l16;
        float bv = bias ? bias[col] : 0.0f;
#pragma unroll
        for (int r = 0; r < 8; ++r) {
            float v = acc[j][r] + bv;
            size_t oi = (size_t)(rtop + r) * M + col;
            if (act == 1) {
                v = v / (1.0f + __expf(-v));         // SiLU
                out16[oi] = (_Float16)v;
            } else {
                out32[oi] = v;
            }
        }
    }
}

// =====================================================================
// Kernel 4: softmax over P=8 per (token, group); in-place on (N x 64)
// =====================================================================
__global__ __launch_bounds__(256) void softmax_p_kernel(float* __restrict__ attn)
{
    int t = blockIdx.x * blockDim.x + threadIdx.x;
    if (t >= N_ * G_) return;
    float* a = attn + (size_t)t * P_;
    float v[P_], m = -1e30f;
#pragma unroll
    for (int p = 0; p < P_; ++p) { v[p] = a[p]; m = fmaxf(m, v[p]); }
    float s = 0.0f;
#pragma unroll
    for (int p = 0; p < P_; ++p) { v[p] = __expf(v[p] - m); s += v[p]; }
    float inv = 1.0f / s;
#pragma unroll
    for (int p = 0; p < P_; ++p) a[p] = v[p] * inv;
}

// =====================================================================
// Kernel 5: deformable bilinear sample + aggregate.
// One block per token; wave = group; lane = channel within group (32).
// Each tap reads 32 contiguous channels (128B coalesced per wave).
// =====================================================================
__global__ __launch_bounds__(256) void sample_agg_kernel(
    const float* __restrict__ val,    // (N, C) fp32, channel = g*32+cg
    const float* __restrict__ off,    // (N, G*P*2)
    const float* __restrict__ attn,   // (N, G*P) softmaxed
    const float* __restrict__ anchor, // (N, 2)
    _Float16* __restrict__ agg)       // (N, C) f16
{
    const int n    = blockIdx.x;
    const int g    = threadIdx.x >> 5;
    const int lane = threadIdx.x & 31;
    const int b    = n / HW_;

    const float ax = anchor[(size_t)n * 2 + 0];
    const float ay = anchor[(size_t)n * 2 + 1];
    const float* offn  = off  + (size_t)n * (G_ * P_ * 2) + g * P_ * 2;
    const float* attnn = attn + (size_t)n * (G_ * P_)     + g * P_;
    const float* valb  = val + (size_t)b * HW_ * C_ + g * CG_ + lane;

    float acc = 0.0f;
#pragma unroll
    for (int p = 0; p < P_; ++p) {
        float ox = offn[p * 2 + 0];
        float oy = offn[p * 2 + 1];
        float aw = attnn[p];
        float px = ax * (float)W_ + ox - 0.5f;
        float py = ay * (float)H_ + oy - 0.5f;
        float x0f = floorf(px), y0f = floorf(py);
        float wx1 = px - x0f, wx0 = 1.0f - wx1;
        float wy1 = py - y0f, wy0 = 1.0f - wy1;
        int x0 = (int)x0f, y0 = (int)y0f;

        float s = 0.0f;
#pragma unroll
        for (int q = 0; q < 4; ++q) {
            int xi = x0 + (q & 1);
            int yi = y0 + (q >> 1);
            float wt = ((q & 1) ? wx1 : wx0) * ((q >> 1) ? wy1 : wy0);
            if (xi >= 0 && xi < W_ && yi >= 0 && yi < H_)
                s += wt * valb[(size_t)(yi * W_ + xi) * C_];
        }
        acc += aw * s;
    }
    agg[(size_t)n * C_ + g * CG_ + lane] = (_Float16)acc;
}

// =====================================================================
// Kernel 6: residual + LayerNorm over C=256 (one block per token).
//   mode 0: store y fp32 (N,C) + f16 (N,C)
//   mode 1: store transposed to (B, C, H, W) fp32 (final output)
// =====================================================================
__global__ __launch_bounds__(256) void ln_residual_kernel(
    const float* __restrict__ xa, const float* __restrict__ xb,
    const float* __restrict__ gamma, const float* __restrict__ beta,
    float* __restrict__ o32, _Float16* __restrict__ o16, int mode)
{
    __shared__ float red[256];
    const int n = blockIdx.x;
    const int c = threadIdx.x;

    float x = xa[(size_t)n * C_ + c] + xb[(size_t)n * C_ + c];

    red[c] = x; __syncthreads();
#pragma unroll
    for (int o = 128; o > 0; o >>= 1) {
        if (c < o) red[c] += red[c + o];
        __syncthreads();
    }
    float mean = red[0] * (1.0f / C_);
    __syncthreads();
    float d = x - mean;
    red[c] = d * d; __syncthreads();
#pragma unroll
    for (int o = 128; o > 0; o >>= 1) {
        if (c < o) red[c] += red[c + o];
        __syncthreads();
    }
    float var = red[0] * (1.0f / C_);
    float y = d * rsqrtf(var + LN_EPS) * gamma[c] + beta[c];

    if (mode == 0) {
        o32[(size_t)n * C_ + c] = y;
        o16[(size_t)n * C_ + c] = (_Float16)y;
    } else {
        int b = n / HW_, s = n % HW_;
        o32[((size_t)b * C_ + c) * HW_ + s] = y;
    }
}

// =====================================================================
// Host launch
// =====================================================================
extern "C" void kernel_launch(void* const* d_in, const int* in_sizes, int n_in,
                              void* d_out, int out_size, void* d_ws, size_t ws_size,
                              hipStream_t stream)
{
    const float* feats     = (const float*)d_in[0];
    const float* feats_pos = (const float*)d_in[1];
    const float* anchor    = (const float*)d_in[2];
    const float* value_W   = (const float*)d_in[3];
    const float* value_b   = (const float*)d_in[4];
    const float* off_W     = (const float*)d_in[5];
    const float* off_b     = (const float*)d_in[6];
    const float* attn_W    = (const float*)d_in[7];
    const float* attn_b    = (const float*)d_in[8];
    const float* out_W     = (const float*)d_in[9];
    const float* out_b     = (const float*)d_in[10];
    const float* ln1_g     = (const float*)d_in[11];
    const float* ln1_b     = (const float*)d_in[12];
    const float* ln2_g     = (const float*)d_in[13];
    const float* ln2_b     = (const float*)d_in[14];
    const float* ffn_w1    = (const float*)d_in[15];
    const float* ffn_b1    = (const float*)d_in[16];
    const float* ffn_w2    = (const float*)d_in[17];
    const float* ffn_b2    = (const float*)d_in[18];
    float* out = (float*)d_out;

    char* ws = (char*)d_ws;
    size_t cur = 0;
    auto alloc = [&](size_t bytes) -> char* {
        char* p = ws + cur;
        cur += (bytes + 255) & ~(size_t)255;
        return p;
    };

    float*    q32    = (float*)   alloc((size_t)N_ * C_ * 4);
    _Float16* q16    = (_Float16*)alloc((size_t)N_ * C_ * 2);   // reused as agg16
    float*    val32  = (float*)   alloc((size_t)N_ * C_ * 4);   // reused as h16 (N x 512 f16)
    float*    off32  = (float*)   alloc((size_t)N_ * 128 * 4);  // reused as y16
    float*    attn32 = (float*)   alloc((size_t)N_ * 64 * 4);
    float*    upd32  = (float*)   alloc((size_t)N_ * C_ * 4);   // reused as f32
    float*    y32    = (float*)   alloc((size_t)N_ * C_ * 4);
    _Float16* wv     = (_Float16*)alloc((size_t)C_ * C_ * 2);
    _Float16* woff   = (_Float16*)alloc((size_t)C_ * 128 * 2);
    _Float16* wattn  = (_Float16*)alloc((size_t)C_ * 64 * 2);
    _Float16* wout   = (_Float16*)alloc((size_t)C_ * C_ * 2);
    _Float16* w1t    = (_Float16*)alloc((size_t)C_ * EC_ * 2);
    _Float16* w2t    = (_Float16*)alloc((size_t)EC_ * C_ * 2);

    _Float16* agg16 = q16;                  // q16 dead after the first 3 GEMMs
    _Float16* h16   = (_Float16*)val32;     // val dead after sampler (same byte size)
    _Float16* y16   = (_Float16*)off32;     // off dead after sampler
    float*    f32   = upd32;                // upd dead after LN1

    // 1) q = feats + feats_pos, NCHW -> (N, C)
    {
        dim3 g(HW_ / 32, C_ / 32, B_), b(32, 8);
        prep_q_kernel<<<g, b, 0, stream>>>(feats, feats_pos, q32, q16);
    }

    // 2) pack weights into WMMA B tiles
    auto pack = [&](const float* W, _Float16* dst, int K, int M, int tr) {
        int total = (K >> 5) * (M >> 4) * 32;
        pack_w_kernel<<<(total + 255) / 256, 256, 0, stream>>>(W, dst, K, M, tr);
    };
    pack(value_W, wv,    C_, C_,  0);
    pack(off_W,   woff,  C_, 128, 0);
    pack(attn_W,  wattn, C_, 64,  0);
    pack(out_W,   wout,  C_, C_,  0);
    pack(ffn_w1,  w1t,   C_, EC_, 1);   // B[k,m] = ffn_w1[m,k]
    pack(ffn_w2,  w2t,   EC_, C_, 1);   // B[k,m] = ffn_w2[m,k]

    auto gemm = [&](const _Float16* A, const _Float16* Bp, const float* bias,
                    float* o32, _Float16* o16, int K, int M, int act) {
        dim3 g(N_ / GEMM_BM, M / GEMM_BN);
        gemm_f16_kernel<<<g, 256, 0, stream>>>(A, Bp, bias, o32, o16, N_, K, M, act);
    };

    // 3-5) value / offset / attention projections
    gemm(q16, wv,    value_b, val32,  nullptr, C_, C_,  0);
    gemm(q16, woff,  off_b,   off32,  nullptr, C_, 128, 0);
    gemm(q16, wattn, attn_b,  attn32, nullptr, C_, 64,  0);

    // 6) softmax over P
    softmax_p_kernel<<<(N_ * G_ + 255) / 256, 256, 0, stream>>>(attn32);

    // 7) deformable sampling + aggregation
    sample_agg_kernel<<<N_, 256, 0, stream>>>(val32, off32, attn32, anchor, agg16);

    // 8) output projection
    gemm(agg16, wout, out_b, upd32, nullptr, C_, C_, 0);

    // 9) y = LN(q + upd)
    ln_residual_kernel<<<N_, 256, 0, stream>>>(q32, upd32, ln1_g, ln1_b,
                                               y32, y16, 0);

    // 10) h = silu(y @ w1^T + b1)  (f16 out)
    gemm(y16, w1t, ffn_b1, nullptr, h16, C_, EC_, 1);

    // 11) f = h @ w2^T + b2
    gemm(h16, w2t, ffn_b2, f32, nullptr, EC_, C_, 0);

    // 12) out = LN(y + f), transposed to (B, C, H, W)
    ln_residual_kernel<<<N_, 256, 0, stream>>>(y32, f32, ln2_g, ln2_b,
                                               out, nullptr, 1);
}
